// T5Attention_47141561041102
// MI455X (gfx1250) — matrix operand order repaired and checked by reading, not compiled, not run on hardware
//
#include <hip/hip_runtime.h>

// ---------------------------------------------------------------------------
// T5 attention (B=2,H=16,S=2048,D=64) flash-attention style for gfx1250.
// bf16 WMMA (v_wmma_f32_16x16x32_bf16) with f32 accumulate, online softmax.
//  - 64-key blocks, bias folded into WMMA C operand (base-2 domain)
//  - row sums computed by WMMA via a constant ones-column B tile
//  - P staged to LDS with packed b64 stores using a K-permutation that is
//    baked into the V-transpose pre-pass
//  - single-base bias gather (constant ds offsets), raw v_exp_f32,
//    hoisted B-tile loads for load/WMMA overlap
// ---------------------------------------------------------------------------

typedef __bf16 bf16_t;
typedef __attribute__((ext_vector_type(4)))  __bf16 v4bf;
typedef __attribute__((ext_vector_type(8)))  __bf16 v8bf;
typedef __attribute__((ext_vector_type(16))) __bf16 v16bf;
typedef __attribute__((ext_vector_type(4)))  float  v4f;
typedef __attribute__((ext_vector_type(8)))  float  v8f;

#define B_ 2
#define H_ 16
#define S_ 2048
#define D_ 64
#define LOG2E_ 1.4426950408889634f
#define SCALEQ_ (0.125f * LOG2E_)   // 1/sqrt(64) * log2(e), folded into Q
#define NEGINF2_ (-30000.0f)        // masked logit in log2 domain
#define PSTR 72                     // P row stride (bf16): 144B, 16B-aligned
#define LN16_ 2.772588722239781f

#define WMMA_BF16(A, Bm, C) \
  __builtin_amdgcn_wmma_f32_16x16x32_bf16(false, (A), false, (Bm), (short)0, \
                                          (C), false, false)
#define EXP2F_(x) __builtin_amdgcn_exp2f(x)   // bare v_exp_f32

// ---------------- pre-pass: fp32 -> bf16 (4 elems / thread) ----------------
__global__ void t5_cvt_bf16(const float* __restrict__ src,
                            bf16_t* __restrict__ dst, float mul, int n4) {
  int i = blockIdx.x * blockDim.x + threadIdx.x;
  if (i < n4) {
    v4f v = *(const v4f*)(src + 4 * i);
    v4bf o;
    o[0] = (bf16_t)(v[0] * mul); o[1] = (bf16_t)(v[1] * mul);
    o[2] = (bf16_t)(v[2] * mul); o[3] = (bf16_t)(v[3] * mul);
    *(v4bf*)(dst + 4 * i) = o;
  }
}

// ---- pre-pass: V (bh,s,d) fp32 -> Vt (bh,d,t') bf16 with K-permutation ----
// Within each 64-key block: t' = c*4 + j  <->  t = 16*j + c  (j=t'&3, c=t'>>2)
__global__ void t5_transpose_v_perm(const float* __restrict__ v,
                                    bf16_t* __restrict__ vt) {
  long idx = (long)blockIdx.x * blockDim.x + threadIdx.x; // over B*H*D*S
  int  tp  = (int)(idx % S_);
  long r   = idx / S_;
  int  d   = (int)(r % D_);
  long bh  = r / D_;
  int  t0  = tp & ~63;
  int  k   = tp & 63;
  int  t   = t0 + 16 * (k & 3) + (k >> 2);
  vt[idx] = (bf16_t)v[(bh * S_ + t) * D_ + d];
}

// ---------------------------- main attention -------------------------------
__launch_bounds__(256, 1)
__global__ void t5_attn_kernel(const bf16_t* __restrict__ Qb,
                               const bf16_t* __restrict__ Kb,
                               const bf16_t* __restrict__ Vt,
                               const float*  __restrict__ bias_table,
                               const long long* __restrict__ event_length,
                               float* __restrict__ out) {
  __shared__ float  s_bias[4096];                     // log2e * bias vs (s-t)
  __shared__ __align__(16) bf16_t s_p[8 * 16 * PSTR]; // per-wave P staging

  const int h    = blockIdx.y;
  const int b    = blockIdx.z;
  const int wave = threadIdx.x >> 5;
  const int lane = threadIdx.x & 31;
  const int q0   = blockIdx.x * 128 + wave * 16;

  // --- one-time per-block: bias lookup table over relative distance -------
  for (int i = threadIdx.x; i < 4095; i += 256) {
    int n = i - 2047;                       // n = s - t
    int bucket;
    if (n < 16) {
      bucket = n < 0 ? 0 : n;
    } else {
      float f  = __logf((float)n * 0.0625f) * (16.0f / LN16_);
      int   vi = (int)f;
      bucket   = 16 + (vi < 15 ? vi : 15);
    }
    s_bias[i] = bias_table[bucket * H_ + h] * LOG2E_;
  }
  __syncthreads();

  const int len = (int)event_length[b];
  const int col = lane & 15;             // N index in B/C layouts
  const int hi8 = (lane >> 4) << 3;      // +8 row offset for upper half-wave
  const int kbA = (lane < 16) ? 0 : 8;   // A-matrix K sub-offset
  const int kbB = (lane < 16) ? 0 : 16;  // B-matrix K sub-offset

  // --- Q tile (16 x 64, pre-scaled) as two A matrices ---------------------
  const bf16_t* qrow = Qb + ((long)(b * H_ + h) * S_ + q0 + col) * D_;
  v16bf a0, a1;
  {
    v8bf c0a = *(const v8bf*)(qrow + kbB);
    v8bf c0b = *(const v8bf*)(qrow + 8 + kbB);
    v8bf c1a = *(const v8bf*)(qrow + 32 + kbB);
    v8bf c1b = *(const v8bf*)(qrow + 40 + kbB);
#pragma unroll
    for (int j = 0; j < 8; ++j) {
      a0[j] = c0a[j]; a0[j + 8] = c0b[j];
      a1[j] = c1a[j]; a1[j + 8] = c1b[j];
    }
  }

  // --- constant ones-column B tile (row-sum extractor) --------------------
  v16bf bones;
  {
    bf16_t one0 = (col == 0) ? (bf16_t)1.0f : (bf16_t)0.0f;
#pragma unroll
    for (int j = 0; j < 16; ++j) bones[j] = one0;
  }

  float m[8];
  v8f   acc[5];                 // [0..3]: output cols, [4]: row-sum column
  v8f   vzero = {};
#pragma unroll
  for (int r = 0; r < 8; ++r) m[r] = -1e30f;
#pragma unroll
  for (int d = 0; d < 5; ++d) acc[d] = vzero;

  const bf16_t* kbase = Kb + (long)(b * H_ + h) * S_ * D_;
  const bf16_t* vbase = Vt + (long)(b * H_ + h) * D_ * S_;
  bf16_t*       pbuf  = s_p + wave * 16 * PSTR;

  auto process_block = [&](int t0, bool MASKED) {
    // ---- logits C-init = bias tile: one base pointer, constant offsets ---
    const float* bp = s_bias + (q0 + hi8 + 2047 - 48 - col - t0);
    v8f c[4];
#pragma unroll
    for (int j = 0; j < 4; ++j)
#pragma unroll
      for (int r = 0; r < 8; ++r) c[j][r] = bp[(3 - j) * 16 + r];

    // ---- QK^T : hoisted B-tile loads, then 8 WMMAs -----------------------
    v16bf bk0[4], bk1[4];
#pragma unroll
    for (int j = 0; j < 4; ++j) {
      const bf16_t* kr = kbase + (long)(t0 + 16 * j + col) * D_;
      bk0[j] = *(const v16bf*)(kr + kbB);
      bk1[j] = *(const v16bf*)(kr + 32 + kbB);
    }
#pragma unroll
    for (int j = 0; j < 4; ++j) {
      c[j] = WMMA_BF16(a0, bk0[j], c[j]);
      c[j] = WMMA_BF16(a1, bk1[j], c[j]);
    }
    // ---- key-side padding mask (tail block only) -------------------------
    if (MASKED) {
#pragma unroll
      for (int j = 0; j < 4; ++j) {
        bool bad = (t0 + 16 * j + col) >= len;
#pragma unroll
        for (int r = 0; r < 8; ++r) c[j][r] = bad ? NEGINF2_ : c[j][r];
      }
    }
    // ---- online softmax: max-butterfly, rescale, exp2, pack P ------------
#pragma unroll
    for (int r = 0; r < 8; ++r) {
      float mx = fmaxf(fmaxf(c[0][r], c[1][r]), fmaxf(c[2][r], c[3][r]));
#pragma unroll
      for (int off = 8; off >= 1; off >>= 1)
        mx = fmaxf(mx, __shfl_xor(mx, off, 32));
      float mnew = fmaxf(m[r], mx);
      float fac  = EXP2F_(m[r] - mnew);
      m[r] = mnew;
#pragma unroll
      for (int d = 0; d < 5; ++d) acc[d][r] = acc[d][r] * fac;
      v4bf pk;
#pragma unroll
      for (int j = 0; j < 4; ++j) pk[j] = (bf16_t)EXP2F_(c[j][r] - mnew);
      // K' = col*4 + j : lane's 4 values are contiguous -> one b64 store
      *(v4bf*)(pbuf + (r + hi8) * PSTR + col * 4) = pk;
    }
    // ---- reload P as two A matrices (permuted K' order) ------------------
    v16bf pa0, pa1;
    {
      const bf16_t* pr  = pbuf + col * PSTR;
      v8bf          p0a = *(const v8bf*)(pr + kbB);
      v8bf          p0b = *(const v8bf*)(pr + 8 + kbB);
      v8bf          p1a = *(const v8bf*)(pr + 32 + kbB);
      v8bf          p1b = *(const v8bf*)(pr + 40 + kbB);
#pragma unroll
      for (int j = 0; j < 8; ++j) {
        pa0[j] = p0a[j]; pa0[j + 8] = p0b[j];
        pa1[j] = p1a[j]; pa1[j + 8] = p1b[j];
      }
    }
    // ---- PV (+ row-sum column) : hoisted loads, Vt in K' order -----------
    v16bf vb0[4], vb1[4];
#pragma unroll
    for (int d = 0; d < 4; ++d) {
      const bf16_t* vr = vbase + (long)(d * 16 + col) * S_ + t0;
      vb0[d] = *(const v16bf*)(vr + kbB);
      vb1[d] = *(const v16bf*)(vr + 32 + kbB);
    }
#pragma unroll
    for (int d = 0; d < 4; ++d) {
      acc[d] = WMMA_BF16(pa0, vb0[d], acc[d]);
      acc[d] = WMMA_BF16(pa1, vb1[d], acc[d]);
    }
    acc[4] = WMMA_BF16(pa0, bones, acc[4]);
    acc[4] = WMMA_BF16(pa1, bones, acc[4]);
    // ---- prefetch next key block (global_prefetch_b8) --------------------
    if (t0 + 64 < S_) {
      __builtin_prefetch(kbase + (long)(t0 + 64 + col) * D_, 0, 1);
      __builtin_prefetch(vbase + (long)col * S_ + t0 + 64, 0, 1);
    }
  };

  const int nfull = len >> 6;               // mask-free 64-key blocks
  const int tail  = len - (nfull << 6);
  for (int blk = 0; blk < nfull; ++blk) process_block(blk << 6, false);
  if (tail) process_block(nfull << 6, true);

  // ---- normalize + store fp32 output -------------------------------------
  float* obase = out + (long)(b * H_ + h) * S_ * D_;
#pragma unroll
  for (int r = 0; r < 8; ++r) {
    float lsum = __shfl(acc[4][r], lane & 16, 32);  // row sum lives in col 0
    float inv  = 1.0f / lsum;
    int   sg   = q0 + r + hi8;
#pragma unroll
    for (int d = 0; d < 4; ++d)
      obase[(long)sg * D_ + d * 16 + col] = acc[d][r] * inv;
  }
}

// ---------------------------------------------------------------------------
extern "C" void kernel_launch(void* const* d_in, const int* in_sizes, int n_in,
                              void* d_out, int out_size, void* d_ws,
                              size_t ws_size, hipStream_t stream) {
  (void)in_sizes; (void)n_in; (void)out_size; (void)ws_size;
  const float*     q    = (const float*)d_in[0];
  const float*     k    = (const float*)d_in[1];
  const float*     v    = (const float*)d_in[2];
  const float*     bias = (const float*)d_in[3];
  const long long* ev   = (const long long*)d_in[4];
  float*           out  = (float*)d_out;

  const long N = (long)B_ * H_ * S_ * D_;  // 4,194,304 elems per tensor
  bf16_t* ws = (bf16_t*)d_ws;              // needs 3*N*2 = 24 MB scratch
  bf16_t* Qb = ws;
  bf16_t* Kb = ws + N;
  bf16_t* Vt = ws + 2 * N;

  const int n4 = (int)(N / 4);
  t5_cvt_bf16<<<(n4 + 255) / 256, 256, 0, stream>>>(q, Qb, SCALEQ_, n4);
  t5_cvt_bf16<<<(n4 + 255) / 256, 256, 0, stream>>>(k, Kb, 1.0f, n4);
  t5_transpose_v_perm<<<(int)(N / 256), 256, 0, stream>>>(v, Vt);

  dim3 grid(S_ / 128, H_, B_);             // (16, 16, 2) blocks of 8 waves
  t5_attn_kernel<<<grid, 256, 0, stream>>>(Qb, Kb, Vt, bias, ev, out);
}
